// TopoteinNetModel_12506944766256
// MI455X (gfx1250) — matrix operand order repaired and checked by reading, not compiled, not run on hardware
//
#include <hip/hip_runtime.h>
#include <math.h>

typedef __bf16 bf16;
typedef __attribute__((ext_vector_type(16))) __bf16 v16bf;
typedef __attribute__((ext_vector_type(8)))  __bf16 v8bf;
typedef __attribute__((ext_vector_type(8)))  float   v8f;

constexpr int   kN    = 32768;
constexpr int   kE    = 262144;
constexpr int   kC    = 4096;
constexpr int   kG    = 32;
constexpr int   kNPG  = kN / kG;   // 1024 nodes per graph (contiguous)
constexpr int   kCELL = 8;
constexpr int   kEC   = 16384;     // edge chunk
constexpr float kEPS  = 1e-8f;

__device__ __forceinline__ float siluf(float x) { return x / (1.f + __expf(-x)); }
__device__ __forceinline__ float sigmf(float x) { return 1.f / (1.f + __expf(-x)); }

// ---------------------------------------------------------------------------
// Generic bf16 WMMA GEMM:  Out[M,Nreal] = act(A[M,Kpad] @ W + bias)
// A: row-major bf16, stride lda == Kpad (multiple of 32, zero-padded cols)
// Wp: packed [Kpad/32][Npad][32] bf16 (zero padded in K and N)
// One wave computes a 16x16 tile, block = 4 waves = 64 rows.
// ---------------------------------------------------------------------------
__global__ __launch_bounds__(128) void k_gemm(
    const bf16* __restrict__ A, int lda,
    const bf16* __restrict__ Wp, int ldw /*Npad*/,
    const float* __restrict__ bias,
    float* __restrict__ Out, int ldo,
    bf16* __restrict__ OutBf, int ldob,
    int M, int Kpad, int Nreal, int act)
{
  int wave   = threadIdx.x >> 5;
  int lane   = threadIdx.x & 31;
  int laneHi = lane >> 4;
  int ln     = lane & 15;
  int m0 = blockIdx.x * 64 + wave * 16;
  int n0 = blockIdx.y * 16;
  if (m0 >= M) return;

  v8f acc = {};
  int row = m0 + ln;
  const bf16* ap = A + (size_t)row * lda + (laneHi ? 8 : 0);
  const bf16* wp = Wp + ((size_t)(n0 + ln)) * 32 + (laneHi ? 8 : 0);
  int ktiles = Kpad >> 5;
  for (int kt = 0; kt < ktiles; ++kt) {
    v8bf a_lo = *(const v8bf*)(ap);
    v8bf a_hi = *(const v8bf*)(ap + 16);
    v8bf b_lo = *(const v8bf*)(wp);
    v8bf b_hi = *(const v8bf*)(wp + 16);
    v16bf a, b;
#pragma unroll
    for (int i = 0; i < 8; ++i) {
      a[i] = a_lo[i]; a[8 + i] = a_hi[i];
      b[i] = b_lo[i]; b[8 + i] = b_hi[i];
    }
    acc = __builtin_amdgcn_wmma_f32_16x16x32_bf16(false, a, false, b,
                                                  (short)0, acc, false, false);
    ap += 32;
    wp += (size_t)ldw * 32;
  }
  int col = n0 + ln;
  if (col < Nreal) {
    float bv = bias ? bias[col] : 0.f;
#pragma unroll
    for (int r = 0; r < 8; ++r) {
      int rr = m0 + r + (laneHi ? 8 : 0);
      float v = acc[r] + bv;
      if (act == 1) v = siluf(v);
      Out[(size_t)rr * ldo + col] = v;
      if (OutBf) OutBf[(size_t)rr * ldob + col] = (bf16)v;
    }
  }
}

// Pack f32 [Kreal,Nreal] row-major weight -> bf16 [Kpad/32][Npad][32], zero pad.
__global__ void k_pack_w(const float* __restrict__ W, bf16* __restrict__ Wp,
                         int Kreal, int Nreal, int Kpad, int Npad) {
  int idx = blockIdx.x * 256 + threadIdx.x;
  if (idx >= Kpad * Npad) return;
  int k = idx / Npad, n = idx % Npad;
  float v = (k < Kreal && n < Nreal) ? W[(size_t)k * Nreal + n] : 0.f;
  int kt = k >> 5, j = k & 31;
  Wp[((size_t)kt * Npad + n) * 32 + j] = (bf16)v;
}

// ------------------------------ aux kernels --------------------------------
__global__ void k_batch_mean(const float* __restrict__ pos, float* __restrict__ gmean) {
  __shared__ float rx[256], ry[256], rz[256];
  int g = blockIdx.x, t = threadIdx.x;
  float sx = 0, sy = 0, sz = 0;
  for (int i = t; i < kNPG; i += 256) {
    const float* p = pos + (size_t)(g * kNPG + i) * 3;
    sx += p[0]; sy += p[1]; sz += p[2];
  }
  rx[t] = sx; ry[t] = sy; rz[t] = sz; __syncthreads();
  for (int off = 128; off > 0; off >>= 1) {
    if (t < off) { rx[t] += rx[t + off]; ry[t] += ry[t + off]; rz[t] += rz[t + off]; }
    __syncthreads();
  }
  if (t == 0) {
    gmean[g * 3 + 0] = rx[0] / kNPG;
    gmean[g * 3 + 1] = ry[0] / kNPG;
    gmean[g * 3 + 2] = rz[0] / kNPG;
  }
}

__global__ void k_centralize(const float* __restrict__ pos, const int* __restrict__ bidx,
                             const float* __restrict__ gmean, float* __restrict__ posc) {
  int i = blockIdx.x * 256 + threadIdx.x;
  if (i >= kN) return;
  int g = bidx[i];
  for (int c = 0; c < 3; ++c)
    posc[(size_t)i * 3 + c] = pos[(size_t)i * 3 + c] - gmean[g * 3 + c];
}

__device__ __forceinline__ void make_frame(const float* p1, const float* p2, float* fr) {
  float a0 = p1[0] - p2[0], a1 = p1[1] - p2[1], a2 = p1[2] - p2[2];
  float an = sqrtf(a0 * a0 + a1 * a1 + a2 * a2 + kEPS);
  a0 /= an; a1 /= an; a2 /= an;
  float b0 = p1[1] * p2[2] - p1[2] * p2[1];
  float b1 = p1[2] * p2[0] - p1[0] * p2[2];
  float b2 = p1[0] * p2[1] - p1[1] * p2[0];
  float bn = sqrtf(b0 * b0 + b1 * b1 + b2 * b2 + kEPS);
  b0 /= bn; b1 /= bn; b2 /= bn;
  float c0 = a1 * b2 - a2 * b1, c1 = a2 * b0 - a0 * b2, c2 = a0 * b1 - a1 * b0;
  fr[0] = a0; fr[1] = a1; fr[2] = a2;
  fr[3] = b0; fr[4] = b1; fr[5] = b2;
  fr[6] = c0; fr[7] = c1; fr[8] = c2;
}

__global__ void k_frames_e(const float* __restrict__ posc, const int* __restrict__ src,
                           const int* __restrict__ dst, float* __restrict__ fr_e) {
  int e = blockIdx.x * 256 + threadIdx.x;
  if (e >= kE) return;
  make_frame(posc + (size_t)src[e] * 3, posc + (size_t)dst[e] * 3, fr_e + (size_t)e * 9);
}

__global__ void k_frames_c(const float* __restrict__ posc, float* __restrict__ fr_c) {
  int c = blockIdx.x * 256 + threadIdx.x;
  if (c >= kC) return;
  make_frame(posc + (size_t)(c * kCELL + kCELL - 1) * 3, posc + (size_t)(c * kCELL) * 3,
             fr_c + (size_t)c * 9);
}

__global__ void k_deg(const int* __restrict__ dst, float* __restrict__ deg) {
  int e = blockIdx.x * 256 + threadIdx.x;
  if (e >= kE) return;
  atomicAdd(&deg[dst[e]], 1.f);
}

__global__ void k_cell_mean_x(const float* __restrict__ x, bf16* __restrict__ xp_bf) {
  int idx = blockIdx.x * 256 + threadIdx.x;
  if (idx >= kC * 64) return;
  int c = idx / 64, f = idx % 64;
  float s = 0;
  for (int r = 0; r < kCELL; ++r) s += x[(size_t)(c * kCELL + r) * 64 + f];
  xp_bf[idx] = (bf16)(s / kCELL);
}

__global__ void k_cell_mean_pos(const float* __restrict__ posc, float* __restrict__ vpool) {
  int idx = blockIdx.x * 256 + threadIdx.x;
  if (idx >= kC * 3) return;
  int c = idx / 3, i = idx % 3;
  float s = 0;
  for (int r = 0; r < kCELL; ++r) s += posc[(size_t)(c * kCELL + r) * 3 + i];
  vpool[idx] = s / kCELL;
}

__global__ void k_sse_concat(const float* __restrict__ s0, const float* __restrict__ spr,
                             bf16* __restrict__ out) {
  int idx = blockIdx.x * 256 + threadIdx.x;
  if (idx >= kC * 128) return;
  int c = idx >> 7, f = idx & 127;
  float v = (f < 64) ? s0[(size_t)c * 64 + f] : spr[(size_t)c * 64 + (f - 64)];
  out[idx] = (bf16)v;
}

// rows = (c,i): cols 0..7 = tensorize(vflat)[c,k,i], 8..15 = wv[k]*vpool[c,i], 16..31 = 0
__global__ void k_build_vt(const float* __restrict__ vflat, const float* __restrict__ fr_c,
                           const float* __restrict__ wv, const float* __restrict__ vpool,
                           bf16* __restrict__ vt) {
  int idx = blockIdx.x * 256 + threadIdx.x;
  if (idx >= kC * 3) return;
  int c = idx / 3, i = idx % 3;
  const float* f = fr_c + (size_t)c * 9;
  bf16* o = vt + (size_t)idx * 32;
  for (int k = 0; k < 8; ++k) {
    const float* s = vflat + (size_t)c * 24 + k * 3;
    float vr = s[0] * f[0 * 3 + i] + s[1] * f[1 * 3 + i] + s[2] * f[2 * 3 + i];
    o[k] = (bf16)vr;
  }
  float vp = vpool[(size_t)c * 3 + i];
  for (int k = 0; k < 8; ++k) o[8 + k] = (bf16)(wv[k] * vp);
  for (int k = 0; k < 16; ++k) o[16 + k] = (bf16)0.f;
}

__global__ void k_rho_sc(const float* __restrict__ rhoT, const float* __restrict__ fr_c,
                         float* __restrict__ rho_sc) {
  int idx = blockIdx.x * 256 + threadIdx.x;
  if (idx >= kC * 24) return;
  int c = idx / 24, r = idx % 24, k = r / 3, j = r % 3;
  const float* f = fr_c + (size_t)c * 9 + j * 3;
  float v = 0;
  for (int i = 0; i < 3; ++i) v += rhoT[(size_t)(c * 3 + i) * 8 + k] * f[i];
  rho_sc[idx] = v;
}

__global__ void k_cvt_bf(const float* __restrict__ src, bf16* __restrict__ dst,
                         int M, int Kreal, int Kpad) {
  int idx = blockIdx.x * 256 + threadIdx.x;
  if (idx >= M * Kpad) return;
  int m = idx / Kpad, c = idx % Kpad;
  dst[idx] = (bf16)((c < Kreal) ? src[(size_t)m * Kreal + c] : 0.f);
}

__global__ void k_chi(const float* __restrict__ hv, const float* __restrict__ posc,
                      float* __restrict__ chi) {
  int idx = blockIdx.x * 256 + threadIdx.x;
  if (idx >= kN * 16) return;
  int n = idx >> 4, k = idx & 15;
  const float* p = posc + (size_t)n * 3;
  float inv = 1.f / sqrtf(p[0] * p[0] + p[1] * p[1] + p[2] * p[2] + kEPS);
  float s = hv[(size_t)n * 16 + k];
  float* o = chi + (size_t)n * 48 + k * 3;
  o[0] = s * p[0] * inv; o[1] = s * p[1] * inv; o[2] = s * p[2] * inv;
}

__global__ void k_edge_geom(const float* __restrict__ posc, const int* __restrict__ src,
                            const int* __restrict__ dst, float* __restrict__ dist,
                            float* __restrict__ dirn) {
  int e = blockIdx.x * 256 + threadIdx.x;
  if (e >= kE) return;
  const float* p1 = posc + (size_t)src[e] * 3;
  const float* p2 = posc + (size_t)dst[e] * 3;
  float d0 = p1[0] - p2[0], d1 = p1[1] - p2[1], d2 = p1[2] - p2[2];
  float dn = sqrtf(d0 * d0 + d1 * d1 + d2 * d2 + kEPS);
  dist[e] = dn;
  dirn[(size_t)e * 3 + 0] = d0 / dn;
  dirn[(size_t)e * 3 + 1] = d1 / dn;
  dirn[(size_t)e * 3 + 2] = d2 / dn;
}

__global__ void k_asm_ein(const float* __restrict__ h, const float* __restrict__ dist,
                          const int* __restrict__ src, const int* __restrict__ dst,
                          bf16* __restrict__ out, int base, int count) {
  int idx = blockIdx.x * 256 + threadIdx.x;
  if (idx >= count * 288) return;
  int e = idx / 288, c = idx % 288;
  int ge = base + e;
  float v;
  if (c < 128)      v = h[(size_t)src[ge] * 128 + c];
  else if (c < 256) v = h[(size_t)dst[ge] * 128 + (c - 128)];
  else if (c == 256) v = dist[ge];
  else              v = 0.f;
  out[(size_t)e * 288 + c] = (bf16)v;
}

__global__ void k_xi(const float* __restrict__ ev, const float* __restrict__ dirn,
                     float* __restrict__ xi, int base, int count) {
  int idx = blockIdx.x * 256 + threadIdx.x;
  if (idx >= count * 4) return;
  int e = idx >> 2, v = idx & 3;
  int ge = base + e;
  float s = ev[(size_t)e * 4 + v];
  const float* d = dirn + (size_t)ge * 3;
  float* o = xi + (size_t)ge * 12 + v * 3;
  o[0] = s * d[0]; o[1] = s * d[1]; o[2] = s * d[2];
}

// msg_in layout: [0,128)h[src] [128,256)h[dst] [256,288)e [288,336)sc(chi_src)
// [336,384)sc(chi_dst) [384,396)sc(xi) [396,460)c_s [460,484)rho_sc [484,512)=0
__global__ void k_asm_msg(const float* __restrict__ h, const bf16* __restrict__ ebf,
                          const float* __restrict__ chi, const float* __restrict__ xi,
                          const float* __restrict__ fr_e, const float* __restrict__ c_s,
                          const float* __restrict__ rho_sc, const int* __restrict__ src,
                          const int* __restrict__ dst, const int* __restrict__ cmap,
                          bf16* __restrict__ out, int base, int count) {
  int idx = blockIdx.x * 256 + threadIdx.x;
  if (idx >= count * 512) return;
  int e = idx >> 9, c = idx & 511;
  int ge = base + e;
  float v = 0.f;
  if (c < 128)      v = h[(size_t)src[ge] * 128 + c];
  else if (c < 256) v = h[(size_t)dst[ge] * 128 + (c - 128)];
  else if (c < 288) v = (float)ebf[(size_t)ge * 32 + (c - 256)];
  else if (c < 384) {
    int r = c - 288;
    int node = (r < 48) ? src[ge] : dst[ge];
    if (r >= 48) r -= 48;
    int k = r / 3, j = r % 3;
    const float* cv = chi + (size_t)node * 48 + k * 3;
    const float* f = fr_e + (size_t)ge * 9 + j * 3;
    v = cv[0] * f[0] + cv[1] * f[1] + cv[2] * f[2];
  } else if (c < 396) {
    int r = c - 384, k = r / 3, j = r % 3;
    const float* xv = xi + (size_t)ge * 12 + k * 3;
    const float* f = fr_e + (size_t)ge * 9 + j * 3;
    v = xv[0] * f[0] + xv[1] * f[1] + xv[2] * f[2];
  } else if (c < 460) v = c_s[(size_t)cmap[dst[ge]] * 64 + (c - 396)];
  else if (c < 484)   v = rho_sc[(size_t)cmap[dst[ge]] * 24 + (c - 460)];
  out[(size_t)e * 512 + c] = (bf16)v;
}

__global__ void k_scatter_s(const float* __restrict__ m, const int* __restrict__ dst,
                            float* __restrict__ agg_s, int base, int count) {
  int idx = blockIdx.x * 256 + threadIdx.x;
  if (idx >= count * 128) return;
  int e = idx >> 7, f = idx & 127;
  atomicAdd(&agg_s[(size_t)dst[base + e] * 128 + f], m[(size_t)e * 176 + f]);
}

__global__ void k_scatter_v(const float* __restrict__ m, const float* __restrict__ fr_e,
                            const int* __restrict__ dst, float* __restrict__ agg_v,
                            int base, int count) {
  int idx = blockIdx.x * 256 + threadIdx.x;
  if (idx >= count * 16) return;
  int e = idx >> 4, k = idx & 15;
  int ge = base + e;
  const float* mk = m + (size_t)e * 176 + 128 + k * 3;
  const float* f = fr_e + (size_t)ge * 9;
  float v0 = mk[0] * f[0] + mk[1] * f[3] + mk[2] * f[6];
  float v1 = mk[0] * f[1] + mk[1] * f[4] + mk[2] * f[7];
  float v2 = mk[0] * f[2] + mk[1] * f[5] + mk[2] * f[8];
  float* a = agg_v + (size_t)dst[ge] * 48 + k * 3;
  atomicAdd(a + 0, v0); atomicAdd(a + 1, v1); atomicAdd(a + 2, v2);
}

__global__ void k_asm_uin(const float* __restrict__ h, const float* __restrict__ agg_s,
                          const float* __restrict__ deg, const float* __restrict__ chi,
                          bf16* __restrict__ out) {
  int idx = blockIdx.x * 256 + threadIdx.x;
  if (idx >= kN * 288) return;
  int n = idx / 288, c = idx % 288;
  float v = 0.f;
  if (c < 128)      v = h[(size_t)n * 128 + c];
  else if (c < 256) v = agg_s[(size_t)n * 128 + (c - 128)] / fmaxf(deg[n], 1.f);
  else if (c < 272) {
    const float* cv = chi + (size_t)n * 48 + (c - 256) * 3;
    v = sqrtf(cv[0] * cv[0] + cv[1] * cv[1] + cv[2] * cv[2] + kEPS);
  }
  out[idx] = (bf16)v;
}

__global__ void k_update_h(float* __restrict__ h, const float* __restrict__ u) {
  int idx = blockIdx.x * 256 + threadIdx.x;
  if (idx >= kN * 128) return;
  int n = idx >> 7, f = idx & 127;
  h[idx] += u[(size_t)n * 144 + f];
}

__global__ void k_update_chi(float* __restrict__ chi, const float* __restrict__ u,
                             const float* __restrict__ agg_v, const float* __restrict__ deg) {
  int idx = blockIdx.x * 256 + threadIdx.x;
  if (idx >= kN * 16) return;
  int n = idx >> 4, k = idx & 15;
  float g = sigmf(u[(size_t)n * 144 + 128 + k]);
  float inv = 1.f / fmaxf(deg[n], 1.f);
  float* c = chi + (size_t)n * 48 + k * 3;
  const float* a = agg_v + (size_t)n * 48 + k * 3;
  c[0] = c[0] * g + a[0] * inv;
  c[1] = c[1] * g + a[1] * inv;
  c[2] = c[2] * g + a[2] * inv;
}

__global__ void k_out_asm(const float* __restrict__ h, const float* __restrict__ chi,
                          const float* __restrict__ gamma, const float* __restrict__ beta,
                          bf16* __restrict__ out) {
  int n = blockIdx.x * 256 + threadIdx.x;
  if (n >= kN) return;
  const float* hr = h + (size_t)n * 128;
  float mu = 0;
  for (int f = 0; f < 128; ++f) mu += hr[f];
  mu *= (1.f / 128.f);
  float var = 0;
  for (int f = 0; f < 128; ++f) { float d = hr[f] - mu; var += d * d; }
  var *= (1.f / 128.f);
  float inv = rsqrtf(var + kEPS);
  bf16* o = out + (size_t)n * 160;
  for (int f = 0; f < 128; ++f) o[f] = (bf16)(gamma[f] * (hr[f] - mu) * inv + beta[f]);
  float vn[16], ms = 0;
  for (int k = 0; k < 16; ++k) {
    const float* cv = chi + (size_t)n * 48 + k * 3;
    vn[k] = sqrtf(cv[0] * cv[0] + cv[1] * cv[1] + cv[2] * cv[2] + kEPS);
    ms += vn[k] * vn[k];
  }
  float rms = sqrtf(ms / 16.f + kEPS);
  for (int k = 0; k < 16; ++k) o[128 + k] = (bf16)(vn[k] / rms);
  for (int k = 0; k < 16; ++k) o[144 + k] = (bf16)0.f;
}

__global__ void k_graph_out(const float* __restrict__ out, float* __restrict__ gout) {
  int idx = blockIdx.x * 256 + threadIdx.x;
  if (idx >= kG * 128) return;
  int g = idx >> 7, f = idx & 127;
  float s = 0;
  for (int i = 0; i < kNPG; ++i) s += out[(size_t)(g * kNPG + i) * 128 + f];
  gout[idx] = s / kNPG;
}

// ---------------------------------------------------------------------------
extern "C" void kernel_launch(void* const* d_in, const int* in_sizes, int n_in,
                              void* d_out, int out_size, void* d_ws, size_t ws_size,
                              hipStream_t stream) {
  (void)in_sizes; (void)n_in; (void)out_size; (void)ws_size;
  const float* x    = (const float*)d_in[0];
  const float* pos  = (const float*)d_in[1];
  const int*   ei   = (const int*)d_in[2];
  const int*   src  = ei;
  const int*   dst  = ei + kE;
  const int*   cmap = (const int*)d_in[3];
  const int*   bidx = (const int*)d_in[4];
  const float *Wx1=(const float*)d_in[5],  *bx1=(const float*)d_in[6];
  const float *Wx2=(const float*)d_in[7],  *bx2=(const float*)d_in[8];
  const float *Wchi=(const float*)d_in[9];
  const float *We1=(const float*)d_in[10], *be1=(const float*)d_in[11];
  const float *We2=(const float*)d_in[12], *be2=(const float*)d_in[13];
  const float *Wxi=(const float*)d_in[14];
  const float *Wg1=(const float*)d_in[15], *bg1=(const float*)d_in[16];
  const float *Wg2=(const float*)d_in[17], *bg2=(const float*)d_in[18];
  const float *Wsse1=(const float*)d_in[19], *bsse1=(const float*)d_in[20];
  const float *Wsse2=(const float*)d_in[21], *bsse2=(const float*)d_in[22];
  const float *wv=(const float*)d_in[23];
  const float *Wspr1=(const float*)d_in[24], *bspr1=(const float*)d_in[25];
  const float *Wspr2=(const float*)d_in[26], *bspr2=(const float*)d_in[27];
  const float *Wvpr1=(const float*)d_in[28], *bvpr1=(const float*)d_in[29];
  const float *Wvpr2=(const float*)d_in[30], *bvpr2=(const float*)d_in[31];
  const float *Wsd1=(const float*)d_in[32], *bsd1=(const float*)d_in[33];
  const float *Wsd2=(const float*)d_in[34], *bsd2=(const float*)d_in[35];
  const float *Wvd1=(const float*)d_in[36], *bvd1=(const float*)d_in[37];
  const float *Wvd2=(const float*)d_in[38], *bvd2=(const float*)d_in[39];
  const float *Wm1=(const float*)d_in[40], *bm1=(const float*)d_in[41];
  const float *Wm2=(const float*)d_in[42], *bm2=(const float*)d_in[43];
  const float *Wu1=(const float*)d_in[44], *bu1=(const float*)d_in[45];
  const float *Wu2=(const float*)d_in[46], *bu2=(const float*)d_in[47];
  /* wpos d_in[48] is dead code in the reference (pos never read after update) */
  const float *gamma=(const float*)d_in[49], *beta=(const float*)d_in[50];
  const float *Wout=(const float*)d_in[51], *bout=(const float*)d_in[52];

  char* wsp = (char*)d_ws;
  auto alloc = [&](size_t b) -> char* {
    char* r = wsp; wsp += (b + 255) & ~(size_t)255; return r;
  };
  auto cdiv = [](long a, long b) -> int { return (int)((a + b - 1) / b); };
  auto pack = [&](const float* W, int Kr, int Nr, int Kp, int Np) -> bf16* {
    bf16* o = (bf16*)alloc((size_t)Kp * Np * 2);
    k_pack_w<<<cdiv((long)Kp * Np, 256), 256, 0, stream>>>(W, o, Kr, Nr, Kp, Np);
    return o;
  };
  auto gemm = [&](const bf16* A, int lda, const bf16* Wp, int Npad, const float* bias,
                  float* Out, int ldo, bf16* Obf, int ldob, int M, int Kpad, int Nreal,
                  int act) {
    dim3 grid((M + 63) / 64, (Nreal + 15) / 16);
    k_gemm<<<grid, 128, 0, stream>>>(A, lda, Wp, Npad, bias, Out, ldo, Obf, ldob,
                                     M, Kpad, Nreal, act);
  };

  // ----- packed weights -----
  bf16* Wsse1p = pack(Wsse1, 64, 64, 64, 64);
  bf16* Wsse2p = pack(Wsse2, 64, 64, 64, 64);
  bf16* Wg1p   = pack(Wg1, 64, 128, 64, 128);
  bf16* Wg2p   = pack(Wg2, 128, 128, 128, 128);
  bf16* Wspr1p = pack(Wspr1, 128, 64, 128, 64);
  bf16* Wspr2p = pack(Wspr2, 64, 64, 64, 64);
  bf16* Wvpr1p = pack(Wvpr1, 128, 24, 128, 32);
  bf16* Wvpr2p = pack(Wvpr2, 24, 24, 32, 32);
  bf16* Wvd1p  = pack(Wvd1, 16, 8, 32, 16);
  bf16* Wvd2p  = pack(Wvd2, 8, 8, 32, 16);
  bf16* Wsd1p  = pack(Wsd1, 128, 64, 128, 64);
  bf16* Wsd2p  = pack(Wsd2, 64, 64, 64, 64);
  bf16* Wx1p   = pack(Wx1, 64, 128, 64, 128);
  bf16* Wx2p   = pack(Wx2, 128, 128, 128, 128);
  bf16* Wchip  = pack(Wchi, 128, 16, 128, 16);
  bf16* We1p   = pack(We1, 257, 32, 288, 32);
  bf16* We2p   = pack(We2, 32, 32, 32, 32);
  bf16* Wxip   = pack(Wxi, 32, 4, 32, 16);
  bf16* Woutp  = pack(Wout, 144, 128, 160, 128);
  bf16 *Wm1p[3], *Wm2p[3], *Wu1p[3], *Wu2p[3];
  for (int l = 0; l < 3; ++l) {
    Wm1p[l] = pack(Wm1 + (size_t)l * 484 * 128, 484, 128, 512, 128);
    Wm2p[l] = pack(Wm2 + (size_t)l * 128 * 176, 128, 176, 128, 176);
    Wu1p[l] = pack(Wu1 + (size_t)l * 272 * 128, 272, 128, 288, 128);
    Wu2p[l] = pack(Wu2 + (size_t)l * 128 * 144, 128, 144, 128, 144);
  }

  // ----- workspace buffers -----
  float* posc   = (float*)alloc((size_t)kN * 3 * 4);
  float* gmean  = (float*)alloc((size_t)kG * 3 * 4);
  float* fr_e   = (float*)alloc((size_t)kE * 9 * 4);
  float* fr_c   = (float*)alloc((size_t)kC * 9 * 4);
  float* deg    = (float*)alloc((size_t)kN * 4);
  bf16*  xp_bf  = (bf16*)alloc((size_t)kC * 64 * 2);
  float* t1_f   = (float*)alloc((size_t)kC * 64 * 4);
  bf16*  t1_bf  = (bf16*)alloc((size_t)kC * 64 * 2);
  float* s0_f   = (float*)alloc((size_t)kC * 64 * 4);
  float* g1_f   = (float*)alloc((size_t)kC * 128 * 4);
  bf16*  g1_bf  = (bf16*)alloc((size_t)kC * 128 * 2);
  float* gemb_f = (float*)alloc((size_t)kC * 128 * 4);
  bf16*  gemb_bf= (bf16*)alloc((size_t)kC * 128 * 2);
  float* sprt_f = (float*)alloc((size_t)kC * 64 * 4);
  bf16*  sprt_bf= (bf16*)alloc((size_t)kC * 64 * 2);
  float* spr_f  = (float*)alloc((size_t)kC * 64 * 4);
  bf16*  sses_bf= (bf16*)alloc((size_t)kC * 128 * 2);
  float* vpr1_f = (float*)alloc((size_t)kC * 24 * 4);
  bf16*  vpr1_bf= (bf16*)alloc((size_t)kC * 32 * 2);
  float* vflat_f= (float*)alloc((size_t)kC * 24 * 4);
  float* vpool  = (float*)alloc((size_t)kC * 3 * 4);
  bf16*  vt_bf  = (bf16*)alloc((size_t)kC * 3 * 32 * 2);
  float* vd1_f  = (float*)alloc((size_t)kC * 3 * 8 * 4);
  bf16*  vd1_bf = (bf16*)alloc((size_t)kC * 3 * 32 * 2);
  float* rhoT   = (float*)alloc((size_t)kC * 3 * 8 * 4);
  float* sd1_f  = (float*)alloc((size_t)kC * 64 * 4);
  bf16*  sd1_bf = (bf16*)alloc((size_t)kC * 64 * 2);
  float* c_s    = (float*)alloc((size_t)kC * 64 * 4);
  float* rho_sc = (float*)alloc((size_t)kC * 24 * 4);
  bf16*  x_bf   = (bf16*)alloc((size_t)kN * 64 * 2);
  float* nf1    = (float*)alloc((size_t)kN * 128 * 4);
  bf16*  nb1    = (bf16*)alloc((size_t)kN * 128 * 2);
  float* h      = (float*)alloc((size_t)kN * 128 * 4);
  bf16*  hbf0   = (bf16*)alloc((size_t)kN * 128 * 2);
  float* hv     = (float*)alloc((size_t)kN * 16 * 4);
  float* chi    = (float*)alloc((size_t)kN * 48 * 4);
  float* dist   = (float*)alloc((size_t)kE * 4);
  float* dirn   = (float*)alloc((size_t)kE * 3 * 4);
  bf16*  e_bf   = (bf16*)alloc((size_t)kE * 32 * 2);
  float* xi     = (float*)alloc((size_t)kE * 12 * 4);
  float* agg_s  = (float*)alloc((size_t)kN * 128 * 4);
  float* agg_v  = (float*)alloc((size_t)kN * 48 * 4);
  bf16*  uin_bf = (bf16*)alloc((size_t)kN * 288 * 2);
  float* u_f    = (float*)alloc((size_t)kN * 144 * 4);
  bf16*  oin_bf = (bf16*)alloc((size_t)kN * 160 * 2);
  bf16*  cin_bf = (bf16*)alloc((size_t)kEC * 512 * 2);   // shared edge-chunk input
  float* ch1_f  = (float*)alloc((size_t)kEC * 128 * 4);
  bf16*  ch1_bf = (bf16*)alloc((size_t)kEC * 128 * 2);
  float* cm_f   = (float*)alloc((size_t)kEC * 176 * 4);
  float* ce2_f  = (float*)alloc((size_t)kEC * 32 * 4);
  float* cev_f  = (float*)alloc((size_t)kEC * 4 * 4);

  hipMemsetAsync(deg, 0, (size_t)kN * 4, stream);
  hipMemsetAsync(vpr1_bf, 0, (size_t)kC * 32 * 2, stream);
  hipMemsetAsync(vd1_bf, 0, (size_t)kC * 3 * 32 * 2, stream);

  // ----- geometry / embeddings -----
  k_batch_mean<<<kG, 256, 0, stream>>>(pos, gmean);
  k_centralize<<<cdiv(kN, 256), 256, 0, stream>>>(pos, bidx, gmean, posc);
  k_frames_e<<<cdiv(kE, 256), 256, 0, stream>>>(posc, src, dst, fr_e);
  k_frames_c<<<cdiv(kC, 256), 256, 0, stream>>>(posc, fr_c);
  k_deg<<<cdiv(kE, 256), 256, 0, stream>>>(dst, deg);
  k_cell_mean_x<<<cdiv((long)kC * 64, 256), 256, 0, stream>>>(x, xp_bf);
  k_cell_mean_pos<<<cdiv((long)kC * 3, 256), 256, 0, stream>>>(posc, vpool);

  // SSE / cell chain
  gemm(xp_bf, 64, Wsse1p, 64, bsse1, t1_f, 64, t1_bf, 64, kC, 64, 64, 1);
  gemm(t1_bf, 64, Wsse2p, 64, bsse2, s0_f, 64, nullptr, 0, kC, 64, 64, 1);
  gemm(xp_bf, 64, Wg1p, 128, bg1, g1_f, 128, g1_bf, 128, kC, 64, 128, 1);
  gemm(g1_bf, 128, Wg2p, 128, bg2, gemb_f, 128, gemb_bf, 128, kC, 128, 128, 1);
  gemm(gemb_bf, 128, Wspr1p, 64, bspr1, sprt_f, 64, sprt_bf, 64, kC, 128, 64, 1);
  gemm(sprt_bf, 64, Wspr2p, 64, bspr2, spr_f, 64, nullptr, 0, kC, 64, 64, 1);
  k_sse_concat<<<cdiv((long)kC * 128, 256), 256, 0, stream>>>(s0_f, spr_f, sses_bf);
  gemm(gemb_bf, 128, Wvpr1p, 32, bvpr1, vpr1_f, 24, vpr1_bf, 32, kC, 128, 24, 1);
  gemm(vpr1_bf, 32, Wvpr2p, 32, bvpr2, vflat_f, 24, nullptr, 0, kC, 32, 24, 1);
  k_build_vt<<<cdiv((long)kC * 3, 256), 256, 0, stream>>>(vflat_f, fr_c, wv, vpool, vt_bf);
  gemm(vt_bf, 32, Wvd1p, 16, bvd1, vd1_f, 8, vd1_bf, 32, kC * 3, 32, 8, 1);
  gemm(vd1_bf, 32, Wvd2p, 16, bvd2, rhoT, 8, nullptr, 0, kC * 3, 32, 8, 1);
  gemm(sses_bf, 128, Wsd1p, 64, bsd1, sd1_f, 64, sd1_bf, 64, kC, 128, 64, 1);
  gemm(sd1_bf, 64, Wsd2p, 64, bsd2, c_s, 64, nullptr, 0, kC, 64, 64, 1);
  k_rho_sc<<<cdiv((long)kC * 24, 256), 256, 0, stream>>>(rhoT, fr_c, rho_sc);

  // node embeddings
  k_cvt_bf<<<cdiv((long)kN * 64, 256), 256, 0, stream>>>(x, x_bf, kN, 64, 64);
  gemm(x_bf, 64, Wx1p, 128, bx1, nf1, 128, nb1, 128, kN, 64, 128, 1);
  gemm(nb1, 128, Wx2p, 128, bx2, h, 128, hbf0, 128, kN, 128, 128, 1);
  gemm(hbf0, 128, Wchip, 16, nullptr, hv, 16, nullptr, 0, kN, 128, 16, 0);
  k_chi<<<cdiv((long)kN * 16, 256), 256, 0, stream>>>(hv, posc, chi);
  k_edge_geom<<<cdiv(kE, 256), 256, 0, stream>>>(posc, src, dst, dist, dirn);

  // edge embeddings (chunked)
  for (int cb = 0; cb < kE; cb += kEC) {
    k_asm_ein<<<cdiv((long)kEC * 288, 256), 256, 0, stream>>>(h, dist, src, dst, cin_bf, cb, kEC);
    gemm(cin_bf, 288, We1p, 32, be1, ch1_f, 32, ch1_bf, 32, kEC, 288, 32, 1);
    gemm(ch1_bf, 32, We2p, 32, be2, ce2_f, 32, e_bf + (size_t)cb * 32, 32, kEC, 32, 32, 1);
    gemm(e_bf + (size_t)cb * 32, 32, Wxip, 16, nullptr, cev_f, 4, nullptr, 0, kEC, 32, 4, 0);
    k_xi<<<cdiv((long)kEC * 4, 256), 256, 0, stream>>>(cev_f, dirn, xi, cb, kEC);
  }

  // ----- TCP interaction layers -----
  for (int l = 0; l < 3; ++l) {
    hipMemsetAsync(agg_s, 0, (size_t)kN * 128 * 4, stream);
    hipMemsetAsync(agg_v, 0, (size_t)kN * 48 * 4, stream);
    for (int cb = 0; cb < kE; cb += kEC) {
      k_asm_msg<<<cdiv((long)kEC * 512, 256), 256, 0, stream>>>(
          h, e_bf, chi, xi, fr_e, c_s, rho_sc, src, dst, cmap, cin_bf, cb, kEC);
      gemm(cin_bf, 512, Wm1p[l], 128, bm1 + l * 128, ch1_f, 128, ch1_bf, 128,
           kEC, 512, 128, 1);
      gemm(ch1_bf, 128, Wm2p[l], 176, bm2 + l * 176, cm_f, 176, nullptr, 0,
           kEC, 128, 176, 0);
      k_scatter_s<<<cdiv((long)kEC * 128, 256), 256, 0, stream>>>(cm_f, dst, agg_s, cb, kEC);
      k_scatter_v<<<cdiv((long)kEC * 16, 256), 256, 0, stream>>>(cm_f, fr_e, dst, agg_v, cb, kEC);
    }
    k_asm_uin<<<cdiv((long)kN * 288, 256), 256, 0, stream>>>(h, agg_s, deg, chi, uin_bf);
    gemm(uin_bf, 288, Wu1p[l], 128, bu1 + l * 128, nf1, 128, nb1, 128, kN, 288, 128, 1);
    gemm(nb1, 128, Wu2p[l], 144, bu2 + l * 144, u_f, 144, nullptr, 0, kN, 128, 144, 0);
    k_update_h<<<cdiv((long)kN * 128, 256), 256, 0, stream>>>(h, u_f);
    k_update_chi<<<cdiv((long)kN * 16, 256), 256, 0, stream>>>(chi, u_f, agg_v, deg);
  }

  // ----- output head -----
  k_out_asm<<<cdiv(kN, 256), 256, 0, stream>>>(h, chi, gamma, beta, oin_bf);
  gemm(oin_bf, 160, Woutp, 128, bout, (float*)d_out, 128, nullptr, 0, kN, 160, 128, 1);
  k_graph_out<<<cdiv((long)kG * 128, 256), 256, 0, stream>>>(
      (const float*)d_out, (float*)d_out + (size_t)kN * 128);
}
